// LSTM_Model_9844065042648
// MI455X (gfx1250) — matrix operand order repaired and checked by reading, not compiled, not run on hardware
//
#include <hip/hip_runtime.h>
#include <hip/hip_bf16.h>
#include <math.h>
#include <stdint.h>

// ---------------------------------------------------------------------------
// 2-layer LSTM (B=512,T=128,F=64,H=1024) + dense-sigmoid head for MI455X.
//
// Persistent kernel, T-loop with grid barriers. Per timestep, per layer:
// one fused GEMM+gate pass. Each wave accumulates the 16x16 (batch,j) patch
// of ALL FOUR gates (i,f,g,o) with v_wmma_f32_16x16x32_bf16 (shared A-frag,
// 4 WMMAs per K-step). The K-loop is software-pipelined AND unrolled by 2
// with alternating fragment register sets, so next-step loads overlap
// current-step WMMAs with no rotation moves. Gate nonlinearities applied in
// registers; c updated in place; h ping-pong buffers. 2 grid barriers/step.
// bf16 weights (~25MB) are L2-resident (192MB L2) -> steady-state GEMMs never
// touch HBM. All workspace accesses via address_space(1) -> global_load/store.
// ---------------------------------------------------------------------------

#define GLOBAL __attribute__((address_space(1)))

typedef __attribute__((ext_vector_type(16))) __bf16        v16bf;
typedef __attribute__((ext_vector_type(8)))  float         v8f;
typedef __attribute__((ext_vector_type(8)))  unsigned int  v8u;

#define NBLOCKS  256
#define NTHREADS 256
#define B_  512
#define T_  128
#define F_  64
#define H_  1024
#define G4  4096   // 4*H

__device__ __forceinline__ float sigmoidf_(float x) {
    return 1.0f / (1.0f + __expf(-x));
}

// ---- grid-wide sense-reversing barrier (counter+gen live in workspace) ----
__device__ __forceinline__ void grid_barrier(unsigned* cnt, unsigned* gen) {
    __threadfence();
    __syncthreads();
    if (threadIdx.x == 0) {
        volatile unsigned* vgen = (volatile unsigned*)gen;
        unsigned g = *vgen;
        unsigned old = atomicAdd(cnt, 1u);
        if (old == NBLOCKS - 1u) {
            *(volatile unsigned*)cnt = 0u;
            __threadfence();
            atomicAdd(gen, 1u);
        } else {
            while (*vgen == g) { __builtin_amdgcn_s_sleep(2); }
        }
    }
    __syncthreads();
}

// ---- WMMA fragment loaders (16-bit layouts per CDNA5 ISA) ------
// A: 16x32 (MxK). lanes 0-15: M=lane, K={0..7,16..23}; lanes 16-31: M=lane-16,
// K={8..15,24..31}. bf16 pairs per VGPR -> u32 loads through addrspace(1).
__device__ __forceinline__ v16bf load_a_frag(const __bf16* A, int lda,
                                             int mBase, int k0) {
    int lane = threadIdx.x & 31;
    int m = mBase + (lane & 15);
    const GLOBAL unsigned int* p = (const GLOBAL unsigned int*)(uintptr_t)
        (A + (size_t)m * lda + k0 + ((lane >= 16) ? 8 : 0));
    v8u u;
#pragma unroll
    for (int j = 0; j < 4; ++j) u[j] = p[j];          // K+{0,2,4,6}
#pragma unroll
    for (int j = 0; j < 4; ++j) u[4 + j] = p[8 + j];  // K+{16,18,20,22}
    return __builtin_bit_cast(v16bf, u);
}

// B: 32x16 (KxN) row-major. lane holds col N=nBase+(lane&15); lanes 0-15:
// K={0..7,16..23}; lanes 16-31: K={8..15,24..31} (same order as A).
__device__ __forceinline__ v16bf load_b_frag(const __bf16* Bm, int ldb,
                                             int k0, int nBase) {
    int lane = threadIdx.x & 31;
    const GLOBAL __bf16* p = (const GLOBAL __bf16*)(uintptr_t)
        (Bm + (size_t)(k0 + ((lane >= 16) ? 8 : 0)) * ldb + nBase + (lane & 15));
    v16bf u;
#pragma unroll
    for (int j = 0; j < 8; ++j) { u[j] = p[(size_t)j * ldb]; u[8 + j] = p[(size_t)(16 + j) * ldb]; }
    return u;
}

// ---- software-pipelined GEMM segment: acc[g] += A @ Bm[:, jB + g*H] -------
// Unrolled by 2 with alternating fragment sets: each half-iteration loads one
// set while the 4 WMMAs consume the other -> latency hidden, no copy moves.
// Requires K/32 even (K = 64 or 1024 here).
__device__ __forceinline__ void gemm_seg_pipelined(v8f (&acc)[4],
                                                   const __bf16* A, int lda,
                                                   int K, const __bf16* Bm,
                                                   int mB, int jB) {
    const int lane = threadIdx.x & 31;
    const int nsteps = K >> 5;     // even, >= 2

    v16bf af0 = load_a_frag(A, lda, mB, 0);
    v16bf b0[4];
#pragma unroll
    for (int g = 0; g < 4; ++g) b0[g] = load_b_frag(Bm, G4, 0, jB + g * H_);

    v16bf af1;
    v16bf b1[4];
    int k = 32;
    for (int i = 0; i < (nsteps >> 1) - 1; ++i) {
        __builtin_prefetch(Bm + (size_t)(k + 64 + lane) * G4 + jB, 0, 1);
        af1 = load_a_frag(A, lda, mB, k);
#pragma unroll
        for (int g = 0; g < 4; ++g) b1[g] = load_b_frag(Bm, G4, k, jB + g * H_);
#pragma unroll
        for (int g = 0; g < 4; ++g)
            acc[g] = __builtin_amdgcn_wmma_f32_16x16x32_bf16(
                false, af0, false, b0[g], (short)0, acc[g], false, false);
        k += 32;

        af0 = load_a_frag(A, lda, mB, k);
#pragma unroll
        for (int g = 0; g < 4; ++g) b0[g] = load_b_frag(Bm, G4, k, jB + g * H_);
#pragma unroll
        for (int g = 0; g < 4; ++g)
            acc[g] = __builtin_amdgcn_wmma_f32_16x16x32_bf16(
                false, af1, false, b1[g], (short)0, acc[g], false, false);
        k += 32;
    }
    // tail: one odd K-step left to load (k == (nsteps-1)*32)
    af1 = load_a_frag(A, lda, mB, k);
#pragma unroll
    for (int g = 0; g < 4; ++g) b1[g] = load_b_frag(Bm, G4, k, jB + g * H_);
#pragma unroll
    for (int g = 0; g < 4; ++g)
        acc[g] = __builtin_amdgcn_wmma_f32_16x16x32_bf16(
            false, af0, false, b0[g], (short)0, acc[g], false, false);
#pragma unroll
    for (int g = 0; g < 4; ++g)
        acc[g] = __builtin_amdgcn_wmma_f32_16x16x32_bf16(
            false, af1, false, b1[g], (short)0, acc[g], false, false);
}

// ---- fused GEMM + LSTM gate step for one layer ----------------------------
//   Zg = A0@B0[:,g*H+j] + A1@B1[:,g*H+j]    for g in {i,f,g,o}
//   c  = sigm(Zf)*c + sigm(Zi)*tanh(Zg);  h = sigm(Zo)*tanh(c)
// Tiles: 64 batch-rows x 32 hidden-cols -> 8*32 = 256 tiles = NBLOCKS.
__device__ void lstm_layer_step(const __bf16* A0, int lda0, int K0,
                                const __bf16* B0,
                                const __bf16* A1, int lda1, int K1,
                                const __bf16* B1,
                                const float* bias, float* c, __bf16* hout) {
    const int wave = threadIdx.x >> 5;
    const int lane = threadIdx.x & 31;
    const int tilesTot = (B_ / 64) * (H_ / 32);   // 256

    for (int tile = blockIdx.x; tile < tilesTot; tile += NBLOCKS) {
        int mB = (tile >> 5) * 64 + (wave >> 1) * 16;  // batch row base
        int jB = (tile & 31) * 32 + (wave & 1) * 16;   // hidden col base

        v8f acc[4];
#pragma unroll
        for (int g = 0; g < 4; ++g)
#pragma unroll
            for (int r = 0; r < 8; ++r) acc[g][r] = 0.0f;

        gemm_seg_pipelined(acc, A0, lda0, K0, B0, mB, jB);  // input proj
        gemm_seg_pipelined(acc, A1, lda1, K1, B1, mB, jB);  // recurrent

        // D layout: VGPR r -> M = mB + r (+8 upper half-wave), N = lane&15.
        int jcol = jB + (lane & 15);
        int mrow = mB + ((lane >= 16) ? 8 : 0);
        float bi  = bias[jcol];
        float bff = bias[jcol + H_];
        float bg  = bias[jcol + 2 * H_];
        float bo  = bias[jcol + 3 * H_];
        GLOBAL float*  cg = (GLOBAL float*)(uintptr_t)c;
        GLOBAL __bf16* hg = (GLOBAL __bf16*)(uintptr_t)hout;
#pragma unroll
        for (int r = 0; r < 8; ++r) {
            int idx = (mrow + r) * H_ + jcol;
            float ig = sigmoidf_(acc[0][r] + bi);
            float fg = sigmoidf_(acc[1][r] + bff);
            float gg = tanhf(acc[2][r] + bg);
            float og = sigmoidf_(acc[3][r] + bo);
            float cv = fg * cg[idx] + ig * gg;
            cg[idx] = cv;
            hg[idx] = (__bf16)(og * tanhf(cv));
        }
    }
}

// ---- persistent whole-model kernel ---------------------------------------
__global__ __launch_bounds__(NTHREADS, 1)
void lstm_model_kernel(const float* __restrict__ x,
                       const float* __restrict__ W1, const float* __restrict__ U1,
                       const float* __restrict__ b1,
                       const float* __restrict__ W2, const float* __restrict__ U2,
                       const float* __restrict__ b2,
                       const float* __restrict__ Wd, const float* __restrict__ bd,
                       float* __restrict__ out, unsigned char* __restrict__ ws) {
    // ---- workspace layout (~42.5 MB) ----
    unsigned* cnt = (unsigned*)ws;
    unsigned* gen = cnt + 1;
    __bf16* xbf  = (__bf16*)(ws + 256);
    __bf16* w1b  = xbf  + (size_t)B_ * T_ * F_;
    __bf16* u1b  = w1b  + (size_t)F_ * G4;
    __bf16* w2b  = u1b  + (size_t)H_ * G4;
    __bf16* u2b  = w2b  + (size_t)H_ * G4;
    __bf16* h1b0 = u2b  + (size_t)H_ * G4;   // ping-pong h buffers
    __bf16* h1b1 = h1b0 + (size_t)B_ * H_;
    __bf16* h2b0 = h1b1 + (size_t)B_ * H_;
    __bf16* h2b1 = h2b0 + (size_t)B_ * H_;
    float*  c1   = (float*)(h2b1 + (size_t)B_ * H_);
    float*  c2   = c1 + (size_t)B_ * H_;

    const int gid = blockIdx.x * NTHREADS + threadIdx.x;
    const int GT  = NBLOCKS * NTHREADS;

    // Phase 0: one-time bf16 conversion of inputs/weights; zero states.
    {
        GLOBAL __bf16* gx = (GLOBAL __bf16*)(uintptr_t)xbf;
        for (long i = gid; i < (long)B_ * T_ * F_; i += GT) gx[i] = (__bf16)x[i];
        GLOBAL __bf16* gw1 = (GLOBAL __bf16*)(uintptr_t)w1b;
        for (long i = gid; i < (long)F_ * G4; i += GT) gw1[i] = (__bf16)W1[i];
        GLOBAL __bf16* gu1 = (GLOBAL __bf16*)(uintptr_t)u1b;
        GLOBAL __bf16* gw2 = (GLOBAL __bf16*)(uintptr_t)w2b;
        GLOBAL __bf16* gu2 = (GLOBAL __bf16*)(uintptr_t)u2b;
        for (long i = gid; i < (long)H_ * G4; i += GT) {
            gu1[i] = (__bf16)U1[i];
            gw2[i] = (__bf16)W2[i];
            gu2[i] = (__bf16)U2[i];
        }
        GLOBAL __bf16* gh1 = (GLOBAL __bf16*)(uintptr_t)h1b0;
        GLOBAL __bf16* gh2 = (GLOBAL __bf16*)(uintptr_t)h2b0;
        GLOBAL float*  gc1 = (GLOBAL float*)(uintptr_t)c1;
        GLOBAL float*  gc2 = (GLOBAL float*)(uintptr_t)c2;
        for (long i = gid; i < (long)B_ * H_; i += GT) {
            gh1[i] = (__bf16)0.0f; gh2[i] = (__bf16)0.0f;
            gc1[i] = 0.0f;         gc2[i] = 0.0f;
        }
    }
    grid_barrier(cnt, gen);

    __bf16* h1buf[2] = { h1b0, h1b1 };
    __bf16* h2buf[2] = { h2b0, h2b1 };

    // Time loop: per step, fused GEMM+gates for each layer, 2 barriers.
    for (int t = 0; t < T_; ++t) {
        int p = t & 1;
        // layer 1: z = x_t@W1 + h1_prev@U1; gates -> h1_new, c1
        lstm_layer_step(xbf + (size_t)t * F_, T_ * F_, F_, w1b,
                        h1buf[p], H_, H_, u1b, b1, c1, h1buf[1 - p]);
        grid_barrier(cnt, gen);
        // layer 2: z = h1_new@W2 + h2_prev@U2; gates -> h2_new, c2
        lstm_layer_step(h1buf[1 - p], H_, H_, w2b,
                        h2buf[p], H_, H_, u2b, b2, c2, h2buf[1 - p]);
        grid_barrier(cnt, gen);
    }

    // Dense sigmoid head: out[b] = sigmoid(h2_last[b,:] @ Wd + bd)   (512x1)
    // T_=128 (even): last write at t=127 went to h2buf[0].
    if (gid < B_) {
        const GLOBAL __bf16* hr =
            (const GLOBAL __bf16*)(uintptr_t)(h2b0 + (size_t)gid * H_);
        float s = 0.0f;
        for (int k = 0; k < H_; ++k) s += (float)hr[k] * Wd[k];
        out[gid] = sigmoidf_(s + bd[0]);
    }
}

// Barrier state must be zeroed before the persistent kernel (ws is poisoned).
__global__ void init_barrier_kernel(unsigned* w) {
    if (threadIdx.x < 2) w[threadIdx.x] = 0u;
}

extern "C" void kernel_launch(void* const* d_in, const int* in_sizes, int n_in,
                              void* d_out, int out_size, void* d_ws, size_t ws_size,
                              hipStream_t stream) {
    (void)in_sizes; (void)n_in; (void)out_size; (void)ws_size;
    const float* x  = (const float*)d_in[0];
    const float* W1 = (const float*)d_in[1];
    const float* U1 = (const float*)d_in[2];
    const float* b1 = (const float*)d_in[3];
    const float* W2 = (const float*)d_in[4];
    const float* U2 = (const float*)d_in[5];
    const float* b2 = (const float*)d_in[6];
    const float* Wd = (const float*)d_in[7];
    const float* bd = (const float*)d_in[8];

    init_barrier_kernel<<<1, 32, 0, stream>>>((unsigned*)d_ws);
    lstm_model_kernel<<<NBLOCKS, NTHREADS, 0, stream>>>(
        x, W1, U1, b1, W2, U2, b2, Wd, bd,
        (float*)d_out, (unsigned char*)d_ws);
}
